// CNN_NODE_ATTEN_BLOCK_19344532701964
// MI455X (gfx1250) — matrix operand order repaired and checked by reading, not compiled, not run on hardware
//
#include <hip/hip_runtime.h>
#include <hip/hip_bf16.h>
#include <stddef.h>

// ---------------- types ----------------
typedef __bf16        v16bf __attribute__((ext_vector_type(16)));
typedef float         v8f   __attribute__((ext_vector_type(8)));
typedef unsigned int  u32x8 __attribute__((ext_vector_type(8)));

// Problem constants (from reference)
#define BB   32
#define NN   32
#define EE   128
#define CIN  64
#define COUT 128
#define HH   128
#define WW   128

// Implicit-GEMM tiling
#define KCHUNKS 18          // 9 taps * (64 cin / 32 per chunk)
#define PACKED_HALVES (8 * KCHUNKS * 32 * 16)   // [cout_tile][chunk][lane][16 halves]
#define PACKED_BYTES  (PACKED_HALVES * 2)       // 147456, multiple of 256

// Conv spatial tile: 4 output rows x 16 output cols per block
#define TROWS 4
#define PROWS (TROWS + 2)   // input rows incl. halo
#define PCOLS 18            // input cols incl. halo
#define PATCHN (PROWS * PCOLS * CIN)   // 6912 elements (= 27 * 256)

__device__ __forceinline__ unsigned short f32_to_bf16(float f) {
    unsigned int u = __builtin_bit_cast(unsigned int, f);
    u += 0x7FFFu + ((u >> 16) & 1u);   // round-to-nearest-even
    return (unsigned short)(u >> 16);
}

// --------------------------------------------------------------------------
// Kernel 1: pack conv_w (f32, OIHW) into bf16 A-fragments in a
// 16x32 A layout with contiguous K (lane half g holds k 16g..16g+15, as B), K tap-major:
//   k' = (kh*3+kw)*64 + cin ; chunk c covers k' in [c*32, c*32+32)
// Lane's 16 halves stored contiguously (32B) -> one vector load per lane.
// --------------------------------------------------------------------------
__global__ __launch_bounds__(256) void pack_weights_kernel(
        const float* __restrict__ conv_w, unsigned short* __restrict__ packedW)
{
    int i = blockIdx.x * 256 + threadIdx.x;
    if (i >= PACKED_HALVES) return;
    int h    = i & 15;
    int lane = (i >> 4) & 31;
    int c    = (i >> 9) % KCHUNKS;
    int tile = i / (16 * 32 * KCHUNKS);

    int m  = lane & 15;
    int g  = lane >> 4;
    int v  = h >> 1;
    int hi = h & 1;
    // A K order used here (same as B): contiguous K, lane half g holds K {16g..16g+15}
    int klocal = 16 * g + 2 * v + hi;

    int cin = (c & 1) * 32 + klocal;
    int tap = c >> 1;
    int kh = tap / 3, kw = tap - kh * 3;
    int cout = tile * 16 + m;

    float w = conv_w[((cout * CIN + cin) * 3 + kh) * 3 + kw];
    packedW[i] = f32_to_bf16(w);
}

// --------------------------------------------------------------------------
// Kernel 2: node attention (tiny: ~0.25 GFLOP) -> out_node[B][N][COUT]
// --------------------------------------------------------------------------
__global__ __launch_bounds__(256) void attn_kernel(
        const float* __restrict__ node_embeds, const float* __restrict__ goal_embed,
        const float* __restrict__ wQ, const float* __restrict__ wK,
        const float* __restrict__ wV, const float* __restrict__ final_w,
        const float* __restrict__ final_b, float* __restrict__ out_node)
{
    __shared__ float Qs[NN * EE];
    __shared__ float Ks[NN * EE];
    __shared__ float Vs[NN * EE];
    __shared__ float Ps[NN * NN];

    int b = blockIdx.x;
    int tid = threadIdx.x;

    for (int i = tid; i < NN * EE; i += 256) {
        int n = i >> 7, e = i & (EE - 1);
        float q = 0.f, k = 0.f, vv = 0.f;
        const float* nrow = node_embeds + n * EE;
        const float* grow = goal_embed + b * EE;
        for (int f = 0; f < EE; ++f) {
            float xv = nrow[f];
            q  += xv * wQ[f * EE + e];
            k  += xv * wK[f * EE + e];
            vv += xv * wV[f * EE + e];
        }
        for (int f = 0; f < EE; ++f) {
            float xv = grow[f];
            q  += xv * wQ[(EE + f) * EE + e];
            k  += xv * wK[(EE + f) * EE + e];
            vv += xv * wV[(EE + f) * EE + e];
        }
        Qs[i] = q; Ks[i] = k; Vs[i] = vv;
    }
    __syncthreads();

    for (int i = tid; i < NN * NN; i += 256) {
        int q = i >> 5, n = i & 31;
        float s = 0.f;
        for (int e = 0; e < EE; ++e) s += Qs[q * EE + e] * Ks[n * EE + e];
        Ps[i] = s * 0.08838834764831845f;   // 1/sqrt(128)
    }
    __syncthreads();

    if (tid < NN) {
        float mx = -3.4e38f;
        for (int n = 0; n < NN; ++n) mx = fmaxf(mx, Ps[tid * NN + n]);
        float sum = 0.f;
        for (int n = 0; n < NN; ++n) {
            float e = __expf(Ps[tid * NN + n] - mx);
            Ps[tid * NN + n] = e;
            sum += e;
        }
        float inv = 1.f / sum;
        for (int n = 0; n < NN; ++n) Ps[tid * NN + n] *= inv;
    }
    __syncthreads();

    for (int i = tid; i < NN * EE; i += 256) {
        int q = i >> 7, e = i & (EE - 1);
        float a = 0.f;
        for (int n = 0; n < NN; ++n) a += Ps[q * NN + n] * Vs[n * EE + e];
        Qs[i] = a;
    }
    __syncthreads();

    for (int i = tid; i < NN * COUT; i += 256) {
        int q = i >> 7, o = i & (COUT - 1);
        float a = 0.f;
        for (int e = 0; e < EE; ++e) a += Qs[q * EE + e] * final_w[o * EE + e];
        out_node[((size_t)b * NN + q) * COUT + o] = a + final_b[o];
    }
}

// --------------------------------------------------------------------------
// Kernel 3: fused conv (implicit GEMM, bf16 WMMA) + bias + node gather.
// Block = (b, 4-row h tile, 16-wide w strip); 8 waves; wave t owns couts
// [16t,16t+16). Per K-chunk: one A-fragment load feeds 4 WMMAs (one per
// output row). Patch staged via GLOBAL_LOAD_ASYNC_TO_LDS_B32 (ASYNCcnt),
// then converted f32->bf16 in LDS once.
// --------------------------------------------------------------------------
__global__ __launch_bounds__(256) void conv_fuse_kernel(
        const float* __restrict__ state, const float* __restrict__ conv_b,
        const unsigned short* __restrict__ packedW,
        const float* __restrict__ out_node,
        const int* __restrict__ game_board, const int* __restrict__ char_to_node,
        float* __restrict__ out)
{
    __shared__ __align__(16) float          patchF[PATCHN];  // f32 staging
    __shared__ __align__(32) unsigned short patchB[PATCHN];  // bf16 [row][col][cin]

    int blk   = blockIdx.x;
    int wtile = blk & 7;
    int htile = (blk >> 3) & 31;
    int b     = blk >> 8;
    int w0    = wtile * 16;
    int h0    = htile * TROWS;
    int tid   = threadIdx.x;

    // ---- Stage 1: async-copy input patch into LDS (zero-filled halo) ----
    // i ordered col-innermost so adjacent lanes fetch adjacent global words.
    for (int i = tid; i < PATCHN; i += 256) {
        int col = i % PCOLS;
        int t   = i / PCOLS;
        int cin = t & (CIN - 1);
        int r   = t >> 6;
        int hin = h0 - 1 + r;
        int win = w0 - 1 + col;
        int lidx = (r * PCOLS + col) * CIN + cin;
        if (hin >= 0 && hin < HH && win >= 0 && win < WW) {
            unsigned long long ga = (unsigned long long)(const void*)
                (state + (((size_t)b * CIN + cin) * HH + hin) * WW + win);
            unsigned int la = (unsigned int)(unsigned long long)(void*)&patchF[lidx];
            asm volatile("global_load_async_to_lds_b32 %0, %1, off"
                         :: "v"(la), "v"(ga) : "memory");
        } else {
            patchF[lidx] = 0.f;
        }
    }
    asm volatile("s_wait_asynccnt 0x0" ::: "memory");
    __syncthreads();

    // ---- Stage 2: f32 -> bf16 conversion pass (LDS resident) ----
    for (int i = tid; i < PATCHN; i += 256)
        patchB[i] = f32_to_bf16(patchF[i]);
    __syncthreads();

    // ---- Stage 3: WMMA main loop ----
    int wave = tid >> 5;        // cout tile
    int lane = tid & 31;
    int g    = lane >> 4;       // lane half
    int n    = lane & 15;       // pixel column within strip

    v8f acc[TROWS] = {};
    const unsigned short* abase = packedW + (size_t)wave * KCHUNKS * 32 * 16;

    for (int c = 0; c < KCHUNKS; ++c) {
        // A fragment: 32 contiguous bytes per lane (L2-resident), reused 4x
        u32x8 ua = *(const u32x8*)(abase + ((size_t)c * 32 + lane) * 16);
        if (c + 1 < KCHUNKS)
            __builtin_prefetch((const void*)(abase + ((size_t)(c + 1) * 32 + lane) * 16), 0, 1);
        v16bf A = __builtin_bit_cast(v16bf, ua);

        int tap  = c >> 1;
        int kh   = tap / 3, kw = tap - kh * 3;
        int cinb = (c & 1) * 32 + g * 16;
        int colb = (n + kw) * CIN + cinb;

#pragma unroll
        for (int t = 0; t < TROWS; ++t) {
            // B fragment: 16 contiguous cin halves from LDS (32B aligned)
            u32x8 ub = *(const u32x8*)&patchB[(kh + t) * (PCOLS * CIN) + colb];
            v16bf Bm = __builtin_bit_cast(v16bf, ub);
            acc[t] = __builtin_amdgcn_wmma_f32_16x16x32_bf16(
                        false, A, false, Bm, (short)0, acc[t], false, false);
        }
    }

    // ---- Epilogue: + bias + gathered node row, store f32 ----
    float bias[8];
#pragma unroll
    for (int r = 0; r < 8; ++r)
        bias[r] = conv_b[wave * 16 + r + 8 * g];

    int w = w0 + n;
#pragma unroll
    for (int t = 0; t < TROWS; ++t) {
        int h = h0 + t;
        int board = game_board[((size_t)b * HH + h) * WW + w];
        bool valid = (board >= 0) && (board < NN);
        int node = valid ? char_to_node[board] : 0;
        const float* onode = out_node + ((size_t)b * NN + node) * COUT;
#pragma unroll
        for (int r = 0; r < 8; ++r) {
            int cout = wave * 16 + r + 8 * g;   // C/D layout: VGPR r -> M = r + 8*lanehalf
            float val = acc[t][r] + bias[r] + (valid ? onode[cout] : 0.f);
            out[(((size_t)b * COUT + cout) * HH + h) * WW + w] = val;
        }
    }
}

// --------------------------------------------------------------------------
extern "C" void kernel_launch(void* const* d_in, const int* in_sizes, int n_in,
                              void* d_out, int out_size, void* d_ws, size_t ws_size,
                              hipStream_t stream) {
    const int*   game_board   = (const int*)  d_in[0];
    const float* state        = (const float*)d_in[1];
    const float* node_embeds  = (const float*)d_in[2];
    const float* goal_embed   = (const float*)d_in[3];
    const int*   char_to_node = (const int*)  d_in[4];
    const float* conv_w       = (const float*)d_in[5];
    const float* conv_b       = (const float*)d_in[6];
    const float* wQ           = (const float*)d_in[7];
    const float* wK           = (const float*)d_in[8];
    const float* wV           = (const float*)d_in[9];
    const float* final_w      = (const float*)d_in[10];
    const float* final_b      = (const float*)d_in[11];
    float* out = (float*)d_out;

    unsigned short* packedW  = (unsigned short*)d_ws;
    float*          out_node = (float*)((char*)d_ws + PACKED_BYTES);

    // 1) pack weights into WMMA A-fragment layout (bf16)
    pack_weights_kernel<<<(PACKED_HALVES + 255) / 256, 256, 0, stream>>>(conv_w, packedW);

    // 2) node attention -> out_node[B][N][COUT]
    attn_kernel<<<BB, 256, 0, stream>>>(node_embeds, goal_embed, wQ, wK, wV,
                                        final_w, final_b, out_node);

    // 3) fused implicit-GEMM conv + bias + gather -> out[B][COUT][H][W]
    conv_fuse_kernel<<<BB * (HH / TROWS) * (WW / 16), 256, 0, stream>>>(
        state, conv_b, packedW, out_node, game_board, char_to_node, out);
}
